// MaskedMultiHeadCrossAttention_21406117003306
// MI455X (gfx1250) — matrix operand order repaired and checked by reading, not compiled, not run on hardware
//
#include <hip/hip_runtime.h>
#include <hip/hip_bf16.h>

// ---------------------------------------------------------------------------
// MaskedMultiHeadCrossAttention w/ RoPE on gfx1250 (MI455X), wave32 + WMMA.
//   1) cvt f32->f16 of x1, x2, wq, wk, wv, wp into workspace
//   2) WMMA GEMM, 32x64 register tile per wave (8 v_wmma per K-step, no spills):
//      Q (+bias+RoPE), K (+bias+RoPE), V (+bias, stored [b,h,d,t]),
//      final projection (f32 -> d_out)
//   3) flash-attention: 1 wave per 16-row tile, online softmax, P re-layout
//      through per-wave LDS slice, 4 accumulator WMMAs vs V^T
// ---------------------------------------------------------------------------

typedef _Float16 h16;
typedef __attribute__((ext_vector_type(8)))  _Float16 v8h;
typedef __attribute__((ext_vector_type(16))) _Float16 v16h;
typedef __attribute__((ext_vector_type(8)))  float    v8f;

#define B_N   2
#define T_N   2048
#define C_N   1024
#define H_N   16
#define D_N   64
#define M_N   (B_N * T_N)   // 4096
#define WIN   64            // SRC_LEN lookahead

__device__ __forceinline__ v8h ld8(const h16* p) { return *(const v8h*)p; }

__device__ __forceinline__ v16h cat8(v8h lo, v8h hi) {
  return __builtin_shufflevector(lo, hi, 0,1,2,3,4,5,6,7,8,9,10,11,12,13,14,15);
}

// --------------------------- f32 -> f16 convert -----------------------------
__global__ void cvt_f16_kernel(const float* __restrict__ in, h16* __restrict__ out, int n4) {
  int i = blockIdx.x * blockDim.x + threadIdx.x;
  if (i < n4) {
    const float4 v = ((const float4*)in)[i];
    h16* o = out + (size_t)i * 4;
    o[0] = (h16)v.x; o[1] = (h16)v.y; o[2] = (h16)v.z; o[3] = (h16)v.w;
  }
}

// ------------------------------- WMMA GEMM ----------------------------------
// out[m,n] = sum_k A[m,k] * W[n,k]  (+ bias[n]) ; 32x64 tile per wave.
// mode 0: Q  -> +bias, RoPE, store f16  qh[b][h][t][d]
// mode 1: K  -> +bias, RoPE, store f16  kh[b][h][t][d]
// mode 2: V  -> +bias,       store f16  vT[b][h][d][t]
// mode 3: O  ->               store f32 out[m][n]
__global__ void __launch_bounds__(128, 1)
wmma_gemm_kernel(const h16* __restrict__ A, const h16* __restrict__ W,
                 const float* __restrict__ bias, void* __restrict__ out,
                 int mode) {
  const int lane = threadIdx.x & 31;
  const int wave = threadIdx.x >> 5;
  const int hi   = lane >> 4;       // 0 or 1
  const int lo   = lane & 15;
  const int m0   = (blockIdx.y * 4 + wave) * 32;  // 32 rows per wave
  const int n0   = blockIdx.x * 64;               // 64 cols per wave

  v8f acc[2][4];
  #pragma unroll
  for (int mi = 0; mi < 2; ++mi)
    #pragma unroll
    for (int ni = 0; ni < 4; ++ni) { v8f z = {}; acc[mi][ni] = z; }

  // single base pointer per stream; row strides become immediate offsets
  const h16* abase = A + (size_t)(m0 + lo) * C_N + hi * 8;   // A: K=kb..kb+7, kb+16..kb+23
  const h16* wbase = W + (size_t)(n0 + lo) * C_N + hi * 8;   // B: K=kb..kb+7, kb+16..kb+23

  #pragma unroll 2
  for (int ks = 0; ks < C_N; ks += 32) {
    __builtin_prefetch(abase + ks + 512, 0, 3);
    __builtin_prefetch(wbase + ks + 512, 0, 3);
    v16h a[2], b[4];
    #pragma unroll
    for (int mi = 0; mi < 2; ++mi) {
      const h16* ap = abase + mi * (16 * C_N) + ks;
      a[mi] = cat8(ld8(ap), ld8(ap + 16));
    }
    #pragma unroll
    for (int ni = 0; ni < 4; ++ni) {
      const h16* bp = wbase + ni * (16 * C_N) + ks;
      b[ni] = cat8(ld8(bp), ld8(bp + 16));
    }
    #pragma unroll
    for (int mi = 0; mi < 2; ++mi)
      #pragma unroll
      for (int ni = 0; ni < 4; ++ni)
        acc[mi][ni] = __builtin_amdgcn_wmma_f32_16x16x32_f16(
            false, a[mi], false, b[ni], (short)0, acc[mi][ni], false, false);
  }

  // ------------------------------ epilogue ----------------------------------
  #pragma unroll
  for (int ni = 0; ni < 4; ++ni) {
    const int n  = n0 + ni * 16 + lo;
    const int hh = n >> 6;
    const int dd = n & 63;
    const float bn = (mode != 3) ? bias[n] : 0.0f;
    const float fr = expf((float)(dd >> 1) * -0.28782313662425572f); // 10000^(-2p/64)

    #pragma unroll
    for (int mi = 0; mi < 2; ++mi) {
      #pragma unroll
      for (int r = 0; r < 8; ++r) {
        const int m  = m0 + mi * 16 + r + hi * 8;
        const int t  = m & (T_N - 1);
        const int bi = m >> 11;
        float val = acc[mi][ni][r] + bn;

        if (mode <= 1) {
          // RoPE: pairs are adjacent columns (dd even/odd) -> partner = lane^1
          float other = __shfl_xor(val, 1, 32);
          float sa, ca;
          sincosf((float)t * fr, &sa, &ca);
          val = ((dd & 1) == 0) ? (val * ca - other * sa)
                                : (other * sa + val * ca);
          // qh/kh: [b][h][t][d]
          ((h16*)out)[(((size_t)(bi * H_N + hh)) * T_N + t) * D_N + dd] = (h16)val;
        } else if (mode == 2) {
          // vT: [b][h][d][t]
          ((h16*)out)[(((size_t)(bi * H_N + hh)) * D_N + dd) * T_N + t] = (h16)val;
        } else {
          ((float*)out)[(size_t)m * C_N + n] = val;
        }
      }
    }
  }
}

// --------------------------- flash attention --------------------------------
// grid (B*H, T/16/4), block 128 (4 waves); wave owns one 16-row tile.
__global__ void attn_flash_kernel(const h16* __restrict__ qh, const h16* __restrict__ kh,
                                  const h16* __restrict__ vT, h16* __restrict__ ah) {
  __shared__ h16 lds[4][16][32];   // per-wave 16x32 P staging (1 KB each)

  const int lane = threadIdx.x & 31;
  const int wave = threadIdx.x >> 5;
  const int hi   = lane >> 4;
  const int lo   = lane & 15;
  const int bh   = blockIdx.x;        // b*16 + h
  const int bi   = bh >> 4;
  const int h    = bh & 15;
  const int i0   = (blockIdx.y * 4 + wave) * 16;

  const size_t qkb = (size_t)bh * (T_N * D_N);
  const size_t vb  = (size_t)bh * (D_N * T_N);

  // Q A-fragments for d = 0..31 and 32..63
  const h16* qrow = qh + qkb + (size_t)(i0 + lo) * D_N + hi * 8;
  const v16h aq0 = cat8(ld8(qrow),      ld8(qrow + 16));
  const v16h aq1 = cat8(ld8(qrow + 32), ld8(qrow + 48));

  float mrow[8], lrow[8];
  v8f acc[4];
  #pragma unroll
  for (int r = 0; r < 8; ++r) { mrow[r] = -1e30f; lrow[r] = 0.0f; }
  #pragma unroll
  for (int dt = 0; dt < 4; ++dt) { v8f z = {}; acc[dt] = z; }

  const int jcap = i0 + 16 + WIN;
  const int jend = jcap < T_N ? jcap : T_N;

  for (int j0 = 0; j0 < jend; j0 += 32) {
    // ---- S = (Q K^T) / 8 for two 16-wide j subtiles ----
    v8f s[2];
    #pragma unroll
    for (int jt = 0; jt < 2; ++jt) {
      const int j = j0 + jt * 16 + lo;                  // B-frag col
      const h16* kp = kh + qkb + (size_t)j * D_N + hi * 8;
      v16h bk0 = cat8(ld8(kp),      ld8(kp + 16));      // d = 0..31
      v16h bk1 = cat8(ld8(kp + 32), ld8(kp + 48));      // d = 32..63
      v8f z = {};
      z = __builtin_amdgcn_wmma_f32_16x16x32_f16(false, aq0, false, bk0, (short)0, z, false, false);
      z = __builtin_amdgcn_wmma_f32_16x16x32_f16(false, aq1, false, bk1, (short)0, z, false, false);
      s[jt] = z;
    }

    // scale + mask (allowed: j <= i + 64)
    #pragma unroll
    for (int jt = 0; jt < 2; ++jt)
      #pragma unroll
      for (int r = 0; r < 8; ++r) {
        const int irow = i0 + r + hi * 8;
        const int jcol = j0 + jt * 16 + lo;
        float x = s[jt][r] * 0.125f;
        s[jt][r] = (jcol > irow + WIN) ? -1e30f : x;
      }

    // ---- online softmax per row (row r lives in a 16-lane group) ----
    #pragma unroll
    for (int r = 0; r < 8; ++r) {
      float v = fmaxf(s[0][r], s[1][r]);
      v = fmaxf(v, __shfl_xor(v, 1, 32));
      v = fmaxf(v, __shfl_xor(v, 2, 32));
      v = fmaxf(v, __shfl_xor(v, 4, 32));
      v = fmaxf(v, __shfl_xor(v, 8, 32));
      const float mnew  = fmaxf(mrow[r], v);
      const float alpha = __expf(mrow[r] - mnew);
      const float p0 = __expf(s[0][r] - mnew);
      const float p1 = __expf(s[1][r] - mnew);
      const int Mr = r + hi * 8;
      lds[wave][Mr][lo]      = (h16)p0;   // C-layout -> [row][jlocal]
      lds[wave][Mr][16 + lo] = (h16)p1;
      float rs = p0 + p1;
      rs += __shfl_xor(rs, 1, 32);
      rs += __shfl_xor(rs, 2, 32);
      rs += __shfl_xor(rs, 4, 32);
      rs += __shfl_xor(rs, 8, 32);
      lrow[r] = lrow[r] * alpha + rs;
      mrow[r] = mnew;
      #pragma unroll
      for (int dt = 0; dt < 4; ++dt) acc[dt][r] *= alpha;
    }

    // ---- P (A-layout) from LDS; O += P @ V^T ----
    const h16* pp = &lds[wave][lo][hi * 8];
    v16h pa = cat8(ld8(pp), ld8(pp + 16));
    #pragma unroll
    for (int dt = 0; dt < 4; ++dt) {
      const h16* vp = vT + vb + (size_t)(dt * 16 + lo) * T_N + j0 + hi * 8;
      v16h bv = cat8(ld8(vp), ld8(vp + 16));
      acc[dt] = __builtin_amdgcn_wmma_f32_16x16x32_f16(false, pa, false, bv, (short)0, acc[dt], false, false);
    }
  }

  // ---- epilogue: O / l, store f16 to ah[b][t][h*64+d] ----
  #pragma unroll
  for (int r = 0; r < 8; ++r) {
    const int t   = i0 + r + hi * 8;
    const float inv = 1.0f / lrow[r];
    #pragma unroll
    for (int dt = 0; dt < 4; ++dt) {
      float o = acc[dt][r] * inv;
      ah[((size_t)(bi * T_N + t)) * C_N + h * D_N + dt * 16 + lo] = (h16)o;
    }
  }
}

// ------------------------------- launcher -----------------------------------
extern "C" void kernel_launch(void* const* d_in, const int* in_sizes, int n_in,
                              void* d_out, int out_size, void* d_ws, size_t ws_size,
                              hipStream_t stream) {
  const float* x1 = (const float*)d_in[0];
  const float* x2 = (const float*)d_in[1];
  const float* wq = (const float*)d_in[2];
  const float* bq = (const float*)d_in[3];
  const float* wk = (const float*)d_in[4];
  const float* bk = (const float*)d_in[5];
  const float* wv = (const float*)d_in[6];
  const float* bv = (const float*)d_in[7];
  const float* wp = (const float*)d_in[8];

  char* ws = (char*)d_ws;
  const size_t szX = (size_t)M_N * C_N * sizeof(h16);  // 8 MB
  const size_t szW = (size_t)C_N * C_N * sizeof(h16);  // 2 MB
  h16* x1h = (h16*)ws;                ws += szX;
  h16* x2h = (h16*)ws;                ws += szX;
  h16* wqh = (h16*)ws;                ws += szW;
  h16* wkh = (h16*)ws;                ws += szW;
  h16* wvh = (h16*)ws;                ws += szW;
  h16* wph = (h16*)ws;                ws += szW;
  h16* qhb = (h16*)ws;                ws += szX;       // [b][h][t][d]
  h16* khb = (h16*)ws;                ws += szX;       // [b][h][t][d]
  h16* vtb = (h16*)ws;                ws += szX;       // [b][h][d][t]
  h16* ahb = (h16*)ws;                ws += szX;       // [b*t][c]

  const int nx4 = (M_N * C_N) / 4;
  const int nw4 = (C_N * C_N) / 4;
  cvt_f16_kernel<<<(nx4 + 255) / 256, 256, 0, stream>>>(x1, x1h, nx4);
  cvt_f16_kernel<<<(nx4 + 255) / 256, 256, 0, stream>>>(x2, x2h, nx4);
  cvt_f16_kernel<<<(nw4 + 255) / 256, 256, 0, stream>>>(wq, wqh, nw4);
  cvt_f16_kernel<<<(nw4 + 255) / 256, 256, 0, stream>>>(wk, wkh, nw4);
  cvt_f16_kernel<<<(nw4 + 255) / 256, 256, 0, stream>>>(wv, wvh, nw4);
  cvt_f16_kernel<<<(nw4 + 255) / 256, 256, 0, stream>>>(wp, wph, nw4);

  // 32x64 tile per wave, 4 waves per block over M: grid (N/64, M/128)
  const dim3 ggrid(C_N / 64, M_N / 128);   // (16, 32)
  wmma_gemm_kernel<<<ggrid, 128, 0, stream>>>(x1h, wqh, bq, qhb, 0);
  wmma_gemm_kernel<<<ggrid, 128, 0, stream>>>(x2h, wkh, bk, khb, 1);
  wmma_gemm_kernel<<<ggrid, 128, 0, stream>>>(x2h, wvh, bv, vtb, 2);

  attn_flash_kernel<<<dim3(B_N * H_N, T_N / 64), 128, 0, stream>>>(qhb, khb, vtb, ahb);

  wmma_gemm_kernel<<<ggrid, 128, 0, stream>>>(ahb, wph, nullptr, d_out, 3);
}
